// MultiHeadAttentionBSM2D_27891517620911
// MI455X (gfx1250) — matrix operand order repaired and checked by reading, not compiled, not run on hardware
//
#include <hip/hip_runtime.h>
#include <hip/hip_bf16.h>
#include <stdint.h>

typedef __bf16 bf16_t;
typedef __attribute__((ext_vector_type(8)))  bf16_t v8bf;
typedef __attribute__((ext_vector_type(16))) bf16_t v16bf;
typedef __attribute__((ext_vector_type(8)))  float  v8f;

static constexpr int B_  = 2;
static constexpr int L_  = 2048;
static constexpr int D_  = 1024;
static constexpr int H_  = 16;
static constexpr int DH_ = 64;
static constexpr int BH_ = B_ * H_;
static constexpr int SSTR = 2053;   // LDS score row stride (odd -> conflict-free banks)

// ---------- helpers ----------
// Native f32 -> bf16 conversion (lets the compiler emit v_cvt_*bf16_f32).
__device__ __forceinline__ bf16_t f2bf(float f) { return (bf16_t)f; }

// A-matrix 16x32 bf16 fragment (M x K), src row-major [16+, ld], at (row0,k0)=base.
// lane<16: row=lane, k in {0..7, 16..23}; lane>=16: row=lane-16, k in {8..15, 24..31}.
__device__ __forceinline__ v16bf load_frag_a_bf16(const bf16_t* __restrict__ base,
                                                  int ld, int lane) {
  int r    = lane & 15;
  int koff = (lane < 16) ? 0 : 8;
  const bf16_t* p = base + (size_t)r * ld + koff;
  v8bf lo = *(const v8bf*)(p);
  v8bf hi = *(const v8bf*)(p + 16);
  v16bf o;
#pragma unroll
  for (int i = 0; i < 8; ++i) { o[i] = lo[i]; o[i + 8] = hi[i]; }
  return o;
}

// B-matrix 32x16 bf16 fragment (K x N). Source is row-major rows of B^T
// (weight/key rows: [n, k], k contiguous), at (n0,k0)=base.
// lane n<16: column n, k in {0..7, 16..23}; lane n+16: column n, k in {8..15, 24..31}.
__device__ __forceinline__ v16bf load_frag_b_bf16(const bf16_t* __restrict__ base,
                                                  int ld, int lane) {
  int n    = lane & 15;
  int koff = (lane < 16) ? 0 : 8;
  const bf16_t* p = base + (size_t)n * ld + koff;
  v8bf lo = *(const v8bf*)(p);
  v8bf hi = *(const v8bf*)(p + 16);
  v16bf o;
#pragma unroll
  for (int i = 0; i < 8; ++i) { o[i] = lo[i]; o[i + 8] = hi[i]; }
  return o;
}

__device__ __forceinline__ v8f wmma_bf16(v16bf a, v16bf b, v8f c) {
  return __builtin_amdgcn_wmma_f32_16x16x32_bf16(false, a, false, b,
                                                 (short)0, c, false, false);
}

// ---------- elementwise fp32 -> bf16 ----------
__global__ void cvt_kernel(const float* __restrict__ src, bf16_t* __restrict__ dst, int n) {
  int i = blockIdx.x * 256 + threadIdx.x;
  if (i < n) dst[i] = f2bf(src[i]);
}

// ---------- LayerNorm: one block per row of [B*L, D] ----------
__global__ void ln_kernel(const float* __restrict__ q, const float* __restrict__ gamma,
                          const float* __restrict__ beta, bf16_t* __restrict__ qn) {
  const int row = blockIdx.x;
  const float* x = q + (size_t)row * D_;
  __shared__ float red[256];
  float v[4];
  float s = 0.f;
#pragma unroll
  for (int i = 0; i < 4; ++i) { v[i] = x[threadIdx.x + 256 * i]; s += v[i]; }
  red[threadIdx.x] = s; __syncthreads();
  for (int off = 128; off > 0; off >>= 1) {
    if ((int)threadIdx.x < off) red[threadIdx.x] += red[threadIdx.x + off];
    __syncthreads();
  }
  const float mu = red[0] * (1.0f / D_);
  __syncthreads();
  float s2 = 0.f;
#pragma unroll
  for (int i = 0; i < 4; ++i) { float d = v[i] - mu; s2 += d * d; }
  red[threadIdx.x] = s2; __syncthreads();
  for (int off = 128; off > 0; off >>= 1) {
    if ((int)threadIdx.x < off) red[threadIdx.x] += red[threadIdx.x + off];
    __syncthreads();
  }
  const float rinv = rsqrtf(red[0] * (1.0f / D_) + 1e-6f);
#pragma unroll
  for (int i = 0; i < 4; ++i) {
    int j = threadIdx.x + 256 * i;
    qn[(size_t)row * D_ + j] = f2bf((v[i] - mu) * rinv * gamma[j] + beta[j]);
  }
}

// ---------- generic bf16 WMMA GEMM: C[M,N] = A[M,K] * W[N,K]^T (+resid) ----------
__global__ void gemm_kernel(const bf16_t* __restrict__ A, const bf16_t* __restrict__ W,
                            float* __restrict__ C, const float* __restrict__ resid,
                            int M, int N, int K) {
  const int lane = threadIdx.x & 31;
  const int t = blockIdx.x * 8 + (threadIdx.x >> 5);
  const int ntN = N >> 4;
  const int mt = t / ntN, nt = t % ntN;
  if (mt >= (M >> 4)) return;
  const bf16_t* Abase = A + (size_t)(mt * 16) * K;
  const bf16_t* Bbase = W + (size_t)(nt * 16) * K;
  v8f acc = {};
  for (int k = 0; k < K; k += 32) {
    v16bf af = load_frag_a_bf16(Abase + k, K, lane);
    v16bf bf = load_frag_b_bf16(Bbase + k, K, lane);
    acc = wmma_bf16(af, bf, acc);
  }
  const int col = nt * 16 + (lane & 15);
  const int rb  = mt * 16 + ((lane >= 16) ? 8 : 0);
#pragma unroll
  for (int r = 0; r < 8; ++r) {
    size_t idx = (size_t)(rb + r) * N + col;
    float val = acc[r];
    if (resid) val += resid[idx];
    C[idx] = val;
  }
}

// ---------- per-head L2 normalize (+ optional transposed copy) ----------
__global__ void l2n_kernel(const float* __restrict__ raw, bf16_t* __restrict__ on,
                           bf16_t* __restrict__ ot) {
  int idx = blockIdx.x * 256 + threadIdx.x;   // over B*L*H vectors
  if (idx >= B_ * L_ * H_) return;
  const int h = idx & (H_ - 1);
  const int l = (idx >> 4) & (L_ - 1);
  const int b = idx >> 15;
  const float* src = raw + ((size_t)b * L_ + l) * D_ + h * DH_;
  float s = 0.f;
#pragma unroll 8
  for (int d = 0; d < DH_; ++d) s += src[d] * src[d];
  const float rn = rsqrtf(s);
  const size_t bh = (size_t)b * H_ + h;
  const size_t nb = (bh * L_ + l) * DH_;
#pragma unroll 8
  for (int d = 0; d < DH_; ++d) {
    bf16_t v = f2bf(src[d] * rn);
    on[nb + d] = v;
    if (ot) ot[(bh * DH_ + d) * L_ + l] = v;
  }
}

// ---------- fused scores + mask + softmax + PV for one (bh, 16-row band) ----------
// LDS: scores [16][SSTR] fp32 (128 KB, fits 320 KB WGP LDS -> 2 blocks/WGP) + 4 KB acc.
// After softmax, bf16 probabilities are compacted IN PLACE over each fp32 row
// (bf16[j] clobbers only fp32[j/2] < every future read index in the ascending
// lane-strided scan; wave32 lockstep + per-iteration compiler barrier keep order).
// grid = BH * (L/16) blocks of 256 threads (8 waves).
__global__ void fused_attn_kernel(const bf16_t* __restrict__ qh,
                                  const bf16_t* __restrict__ qk,   // qih_n [BH,L,DH]
                                  const bf16_t* __restrict__ qt,   // qih_t [BH,DH,L]
                                  const int*   __restrict__ mask,
                                  float*       __restrict__ attn,  // [BH,L,L] fp32
                                  bf16_t*      __restrict__ outh)  // [B,L,D] bf16
{
  extern __shared__ char smem_raw[];
  float* sc      = (float*)smem_raw;          // [16][SSTR]
  float* scratch = sc + 16 * SSTR;            // [4*32*8]

  const int lane = threadIdx.x & 31;
  const int wv   = threadIdx.x >> 5;
  const int bh   = blockIdx.x >> 7;           // / (L/16)
  const int mt   = blockIdx.x & 127;
  const int b    = bh >> 4, h = bh & 15;
  const int col_l = lane & 15;
  const int rbofs = (lane >= 16) ? 8 : 0;

  // ---- phase 1: scores into LDS (A fragments are loop-invariant) ----
  const bf16_t* Abase = qh + ((size_t)bh * L_ + mt * 16) * DH_;
  v16bf a0 = load_frag_a_bf16(Abase,      DH_, lane);
  v16bf a1 = load_frag_a_bf16(Abase + 32, DH_, lane);
  for (int nt = wv; nt < L_ / 16; nt += 8) {
    const bf16_t* Bbase = qk + ((size_t)bh * L_ + nt * 16) * DH_;
    v8f acc = {};
    acc = wmma_bf16(a0, load_frag_b_bf16(Bbase,      DH_, lane), acc);
    acc = wmma_bf16(a1, load_frag_b_bf16(Bbase + 32, DH_, lane), acc);
    const int col = nt * 16 + col_l;
#pragma unroll
    for (int r = 0; r < 8; ++r) {
      const int qrow = rbofs + r;                     // local row 0..15
      const int mv = mask[((size_t)b * L_ + mt * 16 + qrow) * L_ + col];
      sc[qrow * SSTR + col] = (mv == 0) ? -1e9f : acc[r];
    }
  }
  __syncthreads();

  // ---- phase 2: softmax over 16 rows of L_; write fp32 attn once;
  //      compact bf16 probs in place for the PV phase ----
#pragma unroll
  for (int rr = 0; rr < 2; ++rr) {
    const int row = wv * 2 + rr;
    float* srow = sc + row * SSTR;
    float mx = -3.4e38f;
    for (int j = lane; j < L_; j += 32) mx = fmaxf(mx, srow[j]);
#pragma unroll
    for (int off = 16; off > 0; off >>= 1) mx = fmaxf(mx, __shfl_xor(mx, off, 32));
    float s = 0.f;
    for (int j = lane; j < L_; j += 32) { float e = __expf(srow[j] - mx); srow[j] = e; s += e; }
#pragma unroll
    for (int off = 16; off > 0; off >>= 1) s += __shfl_xor(s, off, 32);
    const float rs = 1.0f / s;
    float* grow = attn + ((size_t)bh * L_ + mt * 16 + row) * L_;
    bf16_t* brow = (bf16_t*)srow;               // in-place forward compaction
    for (int j = lane; j < L_; j += 32) {
      float p = srow[j] * rs;
      grow[j] = p;                              // full-precision attn output
      brow[j] = f2bf(p);                        // bf16 prob for PV
      asm volatile("" ::: "memory");            // pin load->store iteration order
    }
  }
  __syncthreads();

  // ---- phase 3: PV from LDS bf16 probs: out[16,64] = P[16,L] @ qih[L,64] ----
  const int dt = wv & 3, kh = wv >> 2;                // 4 d-tiles x 2 k-halves
  const bf16_t* Bb = qt + ((size_t)bh * DH_ + dt * 16) * L_;
  v8f acc = {};
  const int ar   = lane & 15;
  const int akof = (lane < 16) ? 0 : 8;
  const bf16_t* prow = (const bf16_t*)(sc + ar * SSTR);
  const int k0 = kh * (L_ / 2);
  for (int k = k0; k < k0 + L_ / 2; k += 32) {
    const bf16_t* p = prow + k + akof;
    v16bf af;
#pragma unroll
    for (int i = 0; i < 8; ++i) { af[i] = p[i]; af[i + 8] = p[16 + i]; }
    acc = wmma_bf16(af, load_frag_b_bf16(Bb + k, L_, lane), acc);
  }
  if (kh == 1) {
    float* dstw = scratch + (dt * 32 + lane) * 8;
#pragma unroll
    for (int r = 0; r < 8; ++r) dstw[r] = acc[r];
  }
  __syncthreads();
  if (kh == 0) {
    const float* srcw = scratch + (dt * 32 + lane) * 8;
#pragma unroll
    for (int r = 0; r < 8; ++r) {
      const float v = acc[r] + srcw[r];
      const int qrow = rbofs + r;
      outh[((size_t)b * L_ + mt * 16 + qrow) * D_ + h * DH_ + dt * 16 + col_l] = f2bf(v);
    }
  }
}

// ---------- launcher ----------
extern "C" void kernel_launch(void* const* d_in, const int* in_sizes, int n_in,
                              void* d_out, int out_size, void* d_ws, size_t ws_size,
                              hipStream_t stream) {
  (void)in_sizes; (void)n_in; (void)out_size; (void)ws_size;
  const float* q     = (const float*)d_in[0];
  // d_in[1] (k) is unused by the reference
  const float* qi    = (const float*)d_in[2];
  const int*   mask  = (const int*)d_in[3];
  const float* gamma = (const float*)d_in[4];
  const float* beta  = (const float*)d_in[5];
  const float* w_qs  = (const float*)d_in[6];
  const float* w_qis = (const float*)d_in[7];
  const float* w_fc  = (const float*)d_in[8];

  float* out  = (float*)d_out;                  // [B, L, D] fp32
  float* attn = out + (size_t)B_ * L_ * D_;     // [B, H, L, L] fp32

  const size_t BLD = (size_t)B_ * L_ * D_;      // 4M elements
  const size_t DD  = (size_t)D_ * D_;

  uintptr_t w = (uintptr_t)d_ws;
  auto alloc = [&](size_t bytes) -> void* {
    void* p = (void*)w;
    w += (bytes + 255) & ~(size_t)255;
    return p;
  };
  bf16_t* qn_bf   = (bf16_t*)alloc(BLD * 2);
  bf16_t* qi_bf   = (bf16_t*)alloc(BLD * 2);
  bf16_t* wqs_bf  = (bf16_t*)alloc(DD * 2);
  bf16_t* wqis_bf = (bf16_t*)alloc(DD * 2);
  bf16_t* wfc_bf  = (bf16_t*)alloc(DD * 2);
  float*  qh_raw  = (float*)alloc(BLD * 4);
  float*  qih_raw = (float*)alloc(BLD * 4);
  bf16_t* qh_n    = (bf16_t*)alloc(BLD * 2);    // [BH, L, DH]
  bf16_t* qih_n   = (bf16_t*)alloc(BLD * 2);    // [BH, L, DH]
  bf16_t* qih_t   = (bf16_t*)alloc(BLD * 2);    // [BH, DH, L]
  bf16_t* ao_bf   = (bf16_t*)alloc(BLD * 2);    // [B, L, D]

  // 1) fp32 -> bf16 staging
  cvt_kernel<<<(int)((BLD + 255) / 256), 256, 0, stream>>>(qi,    qi_bf,   (int)BLD);
  cvt_kernel<<<(int)((DD  + 255) / 256), 256, 0, stream>>>(w_qs,  wqs_bf,  (int)DD);
  cvt_kernel<<<(int)((DD  + 255) / 256), 256, 0, stream>>>(w_qis, wqis_bf, (int)DD);
  cvt_kernel<<<(int)((DD  + 255) / 256), 256, 0, stream>>>(w_fc,  wfc_bf,  (int)DD);

  // 2) LayerNorm(q)
  ln_kernel<<<B_ * L_, 256, 0, stream>>>(q, gamma, beta, qn_bf);

  // 3) projections (M=4096, N=1024, K=1024)
  const int gemm_blocks = ((B_ * L_ / 16) * (D_ / 16)) / 8;   // 2048
  gemm_kernel<<<gemm_blocks, 256, 0, stream>>>(qn_bf, wqs_bf,  qh_raw,  nullptr,
                                               B_ * L_, D_, D_);
  gemm_kernel<<<gemm_blocks, 256, 0, stream>>>(qi_bf, wqis_bf, qih_raw, nullptr,
                                               B_ * L_, D_, D_);

  // 4) per-head L2 normalize (+ transposed copy of qih for PV B-fragments)
  const int nvec_blocks = (B_ * L_ * H_ + 255) / 256;          // 256
  l2n_kernel<<<nvec_blocks, 256, 0, stream>>>(qh_raw,  qh_n,  (bf16_t*)nullptr);
  l2n_kernel<<<nvec_blocks, 256, 0, stream>>>(qih_raw, qih_n, qih_t);

  // 5) fused scores+mask+softmax+PV; attn touched in HBM exactly once
  const size_t smem_bytes = (size_t)(16 * SSTR + 4 * 32 * 8) * sizeof(float); // ~135.5 KB
  fused_attn_kernel<<<BH_ * (L_ / 16), 256, smem_bytes, stream>>>(
      qh_n, qih_n, qih_t, mask, attn, ao_bf);

  // 6) final FC + residual -> d_out out region (fp32)
  gemm_kernel<<<gemm_blocks, 256, 0, stream>>>(ao_bf, wfc_bf, out, q,
                                               B_ * L_, D_, D_);
}
